// MultiHeadAttention_85040352461311
// MI455X (gfx1250) — matrix operand order repaired and checked by reading, not compiled, not run on hardware
//
#include <hip/hip_runtime.h>
#include <math.h>

#define EMBED 1024
#define HEADS 16
#define HDIM  64
#define BATCH 2
#define SEQ   2048
#define N3E   (3 * EMBED)

typedef __attribute__((ext_vector_type(16))) __bf16        bf16x16;
typedef __attribute__((ext_vector_type(8)))  float         f32x8;
typedef __attribute__((ext_vector_type(4)))  unsigned int  u32x4;

union Frag {
  bf16x16 bf;
  u32x4   u[2];
};

__device__ __forceinline__ unsigned short f2bf(float f) {
  return __builtin_bit_cast(unsigned short, static_cast<__bf16>(f));
}
__device__ __forceinline__ unsigned int pack2bf(float lo, float hi) {
  return (unsigned int)f2bf(lo) | ((unsigned int)f2bf(hi) << 16);
}
__device__ __forceinline__ f32x8 zero8() {
  f32x8 z;
#pragma unroll
  for (int i = 0; i < 8; i++) z[i] = 0.0f;
  return z;
}

// ---------------------------------------------------------------------------
// Kernel 1: QKV projection.  C[4096,3072] = X[4096,1024] @ W[1024,3072] + b
// Block tile 128x128 (8 waves, 4x2; each wave 32x64 = 2x4 WMMA tiles).
// Register-staged double-buffered LDS: one barrier per 32-deep K step.
// Epilogue scatters bf16 Q/K/V as [B,H,S,D]; folds 1/sqrt(D) into Q.
// ---------------------------------------------------------------------------
__global__ __launch_bounds__(256) void qkv_gemm_kernel(
    const float* __restrict__ X, const float* __restrict__ W,
    const float* __restrict__ bias,
    unsigned short* __restrict__ q_ws, unsigned short* __restrict__ k_ws,
    unsigned short* __restrict__ v_ws) {
  __shared__ unsigned short ldsA[2][128][32];   // [m][k] bf16
  __shared__ unsigned short ldsBt[2][128][32];  // [n][k] bf16 (transposed)

  const int tid  = threadIdx.x;
  const int lane = tid & 31;
  const int lh   = lane >> 4;
  const int l16  = lane & 15;
  const int wave = tid >> 5;

  const int mBlock = blockIdx.y * 128;
  const int nBlock = blockIdx.x * 128;
  const int row0 = (wave >> 1) * 32;  // 4 wave-rows
  const int col0 = (wave & 1) * 64;   // 2 wave-cols

  f32x8 acc[2][4];
#pragma unroll
  for (int i = 0; i < 2; i++)
#pragma unroll
    for (int j = 0; j < 4; j++) acc[i][j] = zero8();

  // loaders: A = 128x32 fp32 (2048 pairs), B = 32x128 fp32 (2048 pairs)
  const int a_ku = tid & 15;   // k-pair col
  const int a_m0 = tid >> 4;   // row base, 8 iters of 16
  const int b_nn = tid & 127;  // n col
  const int b_p0 = tid >> 7;   // k-pair parity

  float2 aReg[8];
  float  bReg[8][2];

  auto loadRegs = [&](int kt) {
#pragma unroll
    for (int r = 0; r < 8; r++) {
      int mm = a_m0 + r * 16;
      aReg[r] = *(const float2*)&X[(size_t)(mBlock + mm) * EMBED + kt + a_ku * 2];
      int kk = (b_p0 + r * 2) * 2;
      bReg[r][0] = W[(size_t)(kt + kk) * N3E + nBlock + b_nn];
      bReg[r][1] = W[(size_t)(kt + kk + 1) * N3E + nBlock + b_nn];
    }
  };
  auto storeRegs = [&](int buf) {
#pragma unroll
    for (int r = 0; r < 8; r++) {
      int mm = a_m0 + r * 16;
      *(unsigned int*)&ldsA[buf][mm][a_ku * 2] = pack2bf(aReg[r].x, aReg[r].y);
      int kk = (b_p0 + r * 2) * 2;
      *(unsigned int*)&ldsBt[buf][b_nn][kk] = pack2bf(bReg[r][0], bReg[r][1]);
    }
  };

  loadRegs(0);
  int buf = 0;
  for (int it = 0; it < EMBED / 32; it++) {
    storeRegs(buf);
    __syncthreads();
    if (it + 1 < EMBED / 32) loadRegs((it + 1) * 32);  // prefetch under WMMA

    Frag a[2], b[4];
#pragma unroll
    for (int i = 0; i < 2; i++) {
      const unsigned short* p = &ldsA[buf][row0 + i * 16 + l16][0];
      a[i].u[0] = *(const u32x4*)(p + lh * 8);
      a[i].u[1] = *(const u32x4*)(p + 16 + lh * 8);
    }
#pragma unroll
    for (int j = 0; j < 4; j++) {
      const unsigned short* p = &ldsBt[buf][col0 + j * 16 + l16][0];
      b[j].u[0] = *(const u32x4*)(p + lh * 8);
      b[j].u[1] = *(const u32x4*)(p + 16 + lh * 8);
    }
#pragma unroll
    for (int i = 0; i < 2; i++)
#pragma unroll
      for (int j = 0; j < 4; j++)
        acc[i][j] = __builtin_amdgcn_wmma_f32_16x16x32_bf16(
            false, a[i].bf, false, b[j].bf, (short)0, acc[i][j], false, false);
    buf ^= 1;
  }

  // Epilogue: bias, q-scale, scatter to [B,H,S,D] bf16
#pragma unroll
  for (int i = 0; i < 2; i++) {
#pragma unroll
    for (int j = 0; j < 4; j++) {
      int nG = nBlock + col0 + j * 16 + l16;
      float bv = bias[nG];
      int which = nG >> 10;  // 0:q 1:k 2:v
      int e = nG & (EMBED - 1);
      int h = e >> 6, d = e & 63;
      unsigned short* dst = (which == 0) ? q_ws : ((which == 1) ? k_ws : v_ws);
      float scale = (which == 0) ? 0.125f : 1.0f;  // 1/sqrt(64) into Q
#pragma unroll
      for (int r = 0; r < 8; r++) {
        int mG = mBlock + row0 + i * 16 + r + 8 * lh;  // C row = vgpr + 8*half
        int bb = mG >> 11, s = mG & (SEQ - 1);
        dst[((size_t)(bb * HEADS + h) * SEQ + s) * HDIM + d] =
            f2bf((acc[i][j][r] + bv) * scale);
      }
    }
  }
}

// ---------------------------------------------------------------------------
// Kernel 2: flash attention.  Per block: one (b,h), 128 query rows (16/wave).
// Double-buffered 64-key K/V LDS tiles; online softmax; WMMA QK^T and PV.
// ---------------------------------------------------------------------------
__global__ __launch_bounds__(256) void attn_kernel(
    const unsigned short* __restrict__ q_ws,
    const unsigned short* __restrict__ k_ws,
    const unsigned short* __restrict__ v_ws,
    unsigned short* __restrict__ attn_ws) {
  __shared__ unsigned short ldsK[2][64][64];    // [key][d]
  __shared__ unsigned short ldsVt[2][64][64];   // [d][key]
  __shared__ unsigned short ldsP[8][16][64];    // per-wave P transpose buffer

  const int tid  = threadIdx.x;
  const int lane = tid & 31;
  const int lh   = lane >> 4;
  const int l16  = lane & 15;
  const int wave = tid >> 5;

  const int b  = blockIdx.z;
  const int h  = blockIdx.y;
  const int q0 = blockIdx.x * 128 + wave * 16;

  const unsigned short* Qh = q_ws + (size_t)(b * HEADS + h) * SEQ * HDIM;
  const unsigned short* Kh = k_ws + (size_t)(b * HEADS + h) * SEQ * HDIM;
  const unsigned short* Vh = v_ws + (size_t)(b * HEADS + h) * SEQ * HDIM;

  // Q fragments (A-operand layout), resident for whole kernel
  Frag aQ[2];
  {
    const unsigned short* qp = Qh + (size_t)(q0 + l16) * HDIM;
#pragma unroll
    for (int t = 0; t < 2; t++) {
      aQ[t].u[0] = *(const u32x4*)(qp + t * 32 + lh * 8);
      aQ[t].u[1] = *(const u32x4*)(qp + t * 32 + 16 + lh * 8);
    }
  }

  f32x8 o[4];
#pragma unroll
  for (int nt = 0; nt < 4; nt++) o[nt] = zero8();
  float mrow[8], lrow[8];
#pragma unroll
  for (int r = 0; r < 8; r++) { mrow[r] = -1e30f; lrow[r] = 0.0f; }

  const int c_du = tid & 31;  // d-pair (uint) index
  const int c_k0 = tid >> 5;  // key row base

  unsigned int kReg[8], vReg[8];
  auto loadKV = [&](int kt) {
#pragma unroll
    for (int r = 0; r < 8; r++) {
      int key = c_k0 + r * 8;
      kReg[r] = *(const unsigned int*)(Kh + (size_t)(kt + key) * HDIM + c_du * 2);
      vReg[r] = *(const unsigned int*)(Vh + (size_t)(kt + key) * HDIM + c_du * 2);
    }
  };
  auto storeKV = [&](int buf) {
#pragma unroll
    for (int r = 0; r < 8; r++) {
      int key = c_k0 + r * 8;
      *(unsigned int*)&ldsK[buf][key][c_du * 2] = kReg[r];
      ldsVt[buf][c_du * 2 + 0][key] = (unsigned short)(vReg[r] & 0xFFFFu);
      ldsVt[buf][c_du * 2 + 1][key] = (unsigned short)(vReg[r] >> 16);
    }
  };

  loadKV(0);
  int buf = 0;
  for (int kt = 0; kt < SEQ; kt += 64) {
    storeKV(buf);
    __syncthreads();
    if (kt + 64 < SEQ) loadKV(kt + 64);  // prefetch next tile under compute

    // scores: S = Q K^T  (scale pre-folded into Q)
    f32x8 s[4];
#pragma unroll
    for (int nt = 0; nt < 4; nt++) {
      s[nt] = zero8();
      const unsigned short* kp = &ldsK[buf][nt * 16 + l16][0];
#pragma unroll
      for (int t = 0; t < 2; t++) {
        Frag bK;
        bK.u[0] = *(const u32x4*)(kp + t * 32 + lh * 8);
        bK.u[1] = *(const u32x4*)(kp + t * 32 + 16 + lh * 8);
        s[nt] = __builtin_amdgcn_wmma_f32_16x16x32_bf16(
            false, aQ[t].bf, false, bK.bf, (short)0, s[nt], false, false);
      }
    }

    // online softmax (row r lives in one 16-lane group at vgpr index r)
#pragma unroll
    for (int r = 0; r < 8; r++) {
      float mt = fmaxf(fmaxf(s[0][r], s[1][r]), fmaxf(s[2][r], s[3][r]));
#pragma unroll
      for (int off = 8; off >= 1; off >>= 1)
        mt = fmaxf(mt, __shfl_xor(mt, off, 16));
      float mnew  = fmaxf(mrow[r], mt);
      float alpha = __expf(mrow[r] - mnew);
      mrow[r] = mnew;
      float psum = 0.0f;
#pragma unroll
      for (int nt = 0; nt < 4; nt++) {
        float p = __expf(s[nt][r] - mnew);
        s[nt][r] = p;
        psum += p;
      }
#pragma unroll
      for (int off = 8; off >= 1; off >>= 1)
        psum += __shfl_xor(psum, off, 16);
      lrow[r] = lrow[r] * alpha + psum;
#pragma unroll
      for (int nt = 0; nt < 4; nt++) o[nt][r] *= alpha;
    }

    // transpose P (C layout -> A layout) through per-wave LDS
#pragma unroll
    for (int nt = 0; nt < 4; nt += 2)
#pragma unroll
      for (int r = 0; r < 8; r++)
        ldsP[wave][r + 8 * lh][nt * 16 + l16] = f2bf(s[nt][r]);
#pragma unroll
    for (int nt = 1; nt < 4; nt += 2)
#pragma unroll
      for (int r = 0; r < 8; r++)
        ldsP[wave][r + 8 * lh][nt * 16 + l16] = f2bf(s[nt][r]);

    asm volatile("s_wait_dscnt 0" ::: "memory");  // same-wave LDS RAW fence

    Frag aP[2];
    const unsigned short* pp = &ldsP[wave][l16][0];
#pragma unroll
    for (int t = 0; t < 2; t++) {
      aP[t].u[0] = *(const u32x4*)(pp + t * 32 + lh * 8);
      aP[t].u[1] = *(const u32x4*)(pp + t * 32 + 16 + lh * 8);
    }
    // O += P V
#pragma unroll
    for (int nt = 0; nt < 4; nt++) {
      const unsigned short* vp = &ldsVt[buf][nt * 16 + l16][0];
#pragma unroll
      for (int t = 0; t < 2; t++) {
        Frag bV;
        bV.u[0] = *(const u32x4*)(vp + t * 32 + lh * 8);
        bV.u[1] = *(const u32x4*)(vp + t * 32 + 16 + lh * 8);
        o[nt] = __builtin_amdgcn_wmma_f32_16x16x32_bf16(
            false, aP[t].bf, false, bV.bf, (short)0, o[nt], false, false);
      }
    }
    buf ^= 1;
  }

  // normalize and store attn output as bf16 [B,S,E]
#pragma unroll
  for (int nt = 0; nt < 4; nt++) {
    int d = nt * 16 + l16;
#pragma unroll
    for (int r = 0; r < 8; r++) {
      int srow = q0 + r + 8 * lh;
      attn_ws[((size_t)(b * SEQ + srow) * EMBED) + h * HDIM + d] =
          f2bf(o[nt][r] / lrow[r]);
    }
  }
}

// ---------------------------------------------------------------------------
// Kernel 3: out projection.  out[4096,1024] = attn_bf16 @ W[1024,1024] + b
// Same tiling/pipelining as kernel 1; A already bf16, output fp32.
// ---------------------------------------------------------------------------
__global__ __launch_bounds__(256) void out_gemm_kernel(
    const unsigned short* __restrict__ A, const float* __restrict__ W,
    const float* __restrict__ bias, float* __restrict__ out) {
  __shared__ unsigned short ldsA[2][128][32];
  __shared__ unsigned short ldsBt[2][128][32];

  const int tid  = threadIdx.x;
  const int lane = tid & 31;
  const int lh   = lane >> 4;
  const int l16  = lane & 15;
  const int wave = tid >> 5;

  const int mBlock = blockIdx.y * 128;
  const int nBlock = blockIdx.x * 128;
  const int row0 = (wave >> 1) * 32;
  const int col0 = (wave & 1) * 64;

  f32x8 acc[2][4];
#pragma unroll
  for (int i = 0; i < 2; i++)
#pragma unroll
    for (int j = 0; j < 4; j++) acc[i][j] = zero8();

  const int a_ku = tid & 15;
  const int a_m0 = tid >> 4;
  const int b_nn = tid & 127;
  const int b_p0 = tid >> 7;

  unsigned int aReg[8];
  float        bReg[8][2];

  auto loadRegs = [&](int kt) {
#pragma unroll
    for (int r = 0; r < 8; r++) {
      int mm = a_m0 + r * 16;
      aReg[r] = *(const unsigned int*)&A[(size_t)(mBlock + mm) * EMBED + kt + a_ku * 2];
      int kk = (b_p0 + r * 2) * 2;
      bReg[r][0] = W[(size_t)(kt + kk) * EMBED + nBlock + b_nn];
      bReg[r][1] = W[(size_t)(kt + kk + 1) * EMBED + nBlock + b_nn];
    }
  };
  auto storeRegs = [&](int buf) {
#pragma unroll
    for (int r = 0; r < 8; r++) {
      int mm = a_m0 + r * 16;
      *(unsigned int*)&ldsA[buf][mm][a_ku * 2] = aReg[r];
      int kk = (b_p0 + r * 2) * 2;
      *(unsigned int*)&ldsBt[buf][b_nn][kk] = pack2bf(bReg[r][0], bReg[r][1]);
    }
  };

  loadRegs(0);
  int buf = 0;
  for (int it = 0; it < EMBED / 32; it++) {
    storeRegs(buf);
    __syncthreads();
    if (it + 1 < EMBED / 32) loadRegs((it + 1) * 32);

    Frag a[2], b[4];
#pragma unroll
    for (int i = 0; i < 2; i++) {
      const unsigned short* p = &ldsA[buf][row0 + i * 16 + l16][0];
      a[i].u[0] = *(const u32x4*)(p + lh * 8);
      a[i].u[1] = *(const u32x4*)(p + 16 + lh * 8);
    }
#pragma unroll
    for (int j = 0; j < 4; j++) {
      const unsigned short* p = &ldsBt[buf][col0 + j * 16 + l16][0];
      b[j].u[0] = *(const u32x4*)(p + lh * 8);
      b[j].u[1] = *(const u32x4*)(p + 16 + lh * 8);
    }
#pragma unroll
    for (int i = 0; i < 2; i++)
#pragma unroll
      for (int j = 0; j < 4; j++)
        acc[i][j] = __builtin_amdgcn_wmma_f32_16x16x32_bf16(
            false, a[i].bf, false, b[j].bf, (short)0, acc[i][j], false, false);
    buf ^= 1;
  }

#pragma unroll
  for (int i = 0; i < 2; i++) {
#pragma unroll
    for (int j = 0; j < 4; j++) {
      int nG = nBlock + col0 + j * 16 + l16;
      float bv = bias[nG];
#pragma unroll
      for (int r = 0; r < 8; r++) {
        int mG = mBlock + row0 + i * 16 + r + 8 * lh;
        out[(size_t)mG * EMBED + nG] = acc[i][j][r] + bv;
      }
    }
  }
}

// ---------------------------------------------------------------------------
extern "C" void kernel_launch(void* const* d_in, const int* in_sizes, int n_in,
                              void* d_out, int out_size, void* d_ws,
                              size_t ws_size, hipStream_t stream) {
  const float* X     = (const float*)d_in[0];
  const float* qkv_w = (const float*)d_in[1];
  const float* qkv_b = (const float*)d_in[2];
  const float* out_w = (const float*)d_in[3];
  const float* out_b = (const float*)d_in[4];

  const size_t perMat = (size_t)BATCH * HEADS * SEQ * HDIM;  // 4M elems
  unsigned short* q_ws    = (unsigned short*)d_ws;
  unsigned short* k_ws    = q_ws + perMat;
  unsigned short* v_ws    = k_ws + perMat;
  unsigned short* attn_ws = v_ws + perMat;

  dim3 blk(256);
  qkv_gemm_kernel<<<dim3(N3E / 128, (BATCH * SEQ) / 128), blk, 0, stream>>>(
      X, qkv_w, qkv_b, q_ws, k_ws, v_ws);
  attn_kernel<<<dim3(SEQ / 128, HEADS, BATCH), blk, 0, stream>>>(
      q_ws, k_ws, v_ws, attn_ws);
  out_gemm_kernel<<<dim3(EMBED / 128, (BATCH * SEQ) / 128), blk, 0, stream>>>(
      attn_ws, out_w, out_b, (float*)d_out);
}
